// ROIAlignSpatialTemporalModule_33423435498466
// MI455X (gfx1250) — matrix operand order repaired and checked by reading, not compiled, not run on hardware
//
#include <hip/hip_runtime.h>
#include <hip/hip_bf16.h>
#include <math.h>

// ---------------------------------------------------------------------------
// ROIAlignSpatialTemporalModule for MI455X (gfx1250, wave32, WMMA)
//
// Pipeline:
//   K0: w1 fp32 -> bf16 (L2-resident, 512KB)
//   K1: fused GEMM (x . w1^T) + BN + hardswish + partial mean  [bf16 WMMA,
//       double-buffered LDS A-tiles, branch-free clamped staging]
//   K2: reduce partials -> pooled[B][R] (mean over 196 positions)
//   K3: head: pooled @ w2^T + BN2 + sigmoid -> roi params [B][6]
//   K4: roi_align3d from x, LDS-staged via global_load_async_to_lds_b128
// ---------------------------------------------------------------------------

#define B_   128
#define C_   1024
#define T_   4
#define H_   7
#define W_   7
#define P_   196           // T*H*W
#define R_   256
#define MT_  13            // ceil(196/16) M-tiles per batch
#define OT_  3
#define OH_  5
#define OW_  5
#define BINS_ 75
#define EPS_ 1e-5f

typedef __attribute__((ext_vector_type(16))) __bf16 v16bf;
typedef __attribute__((ext_vector_type(8)))  __bf16 v8bf;
typedef __attribute__((ext_vector_type(2)))  __bf16 v2bf;
typedef __attribute__((ext_vector_type(8)))  float  v8f;
typedef __attribute__((ext_vector_type(4)))  int    v4i;

#if defined(__has_builtin)
#if __has_builtin(__builtin_amdgcn_global_load_async_to_lds_b128)
#define USE_ASYNC_LDS 1
#else
#define USE_ASYNC_LDS 0
#endif
#else
#define USE_ASYNC_LDS 0
#endif

__device__ __forceinline__ void wait_async0() {
#if defined(__has_builtin) && __has_builtin(__builtin_amdgcn_s_wait_asynccnt)
  __builtin_amdgcn_s_wait_asynccnt(0);
#else
  asm volatile("s_wait_asynccnt 0" ::: "memory");
#endif
}

// ---------------------------------------------------------------- K0: w1->bf16
__global__ void k0_convert_w1(const float* __restrict__ w1, __bf16* __restrict__ w1b) {
  int i = blockIdx.x * blockDim.x + threadIdx.x;   // 0 .. 262143
  w1b[i] = (__bf16)w1[i];
}

// Branch-free A-tile staging: x[b, kk..kk+31, p0..p0+15] -> buf[p][k] (bf16).
// Thread t: p = t&15, k = 2*(t>>4). Clamped p index keeps every access
// in-bounds; rows with p >= 196 hold garbage that the epilogue masks out.
__device__ __forceinline__ void stage_a(const float* __restrict__ xb, int kk, int p0,
                                        __bf16 (*buf)[32], int tid) {
  const int p  = tid & 15;
  const int k  = (tid >> 4) * 2;               // 0,2,...,30
  const int gp = min(p0 + p, P_ - 1);
  float f0 = xb[(size_t)(kk + k) * P_ + gp];
  float f1 = xb[(size_t)(kk + k + 1) * P_ + gp];
  v2bf pk;
  pk[0] = (__bf16)f0;
  pk[1] = (__bf16)f1;
  *(v2bf*)&buf[p][k] = pk;                     // one aligned ds_store_b32
}

// ------------------------------------------------- K1: GEMM + BN + hswish + sum
// grid: (MT_, B_), block: 256 (8 waves). Wave w owns N columns [32w, 32w+32).
__global__ void __launch_bounds__(256)
k1_gemm_partial(const float* __restrict__ x, const __bf16* __restrict__ w1b,
                const float* __restrict__ g1, const float* __restrict__ b1,
                const float* __restrict__ m1, const float* __restrict__ v1,
                float* __restrict__ partial) {
  __shared__ __align__(16) __bf16 lds_a[2][16][32];  // double-buffered [p][k]

  const int tile = blockIdx.x;        // 0..12
  const int b    = blockIdx.y;        // 0..127
  const int p0   = tile * 16;
  const int lane = threadIdx.x & 31;
  const int wv   = threadIdx.x >> 5;  // 0..7
  const int n0   = wv * 32;
  const int na   = n0 + (lane & 15);
  const int nb   = na + 16;
  const int khalf = (lane >> 4);      // 0 or 1
  const int m    = lane & 15;
  const int kb   = khalf * 8;

  const float* xb = x + (size_t)b * C_ * P_;

  v8f acc0 = {}; v8f acc1 = {};

  stage_a(xb, 0, p0, lds_a[0], threadIdx.x);
  __syncthreads();

  for (int kk = 0; kk < C_; kk += 32) {
    const int cur = (kk >> 5) & 1;

    // stage next chunk into the other buffer while computing this one
    if (kk + 32 < C_) {
      stage_a(xb, kk + 32, p0, lds_a[cur ^ 1], threadIdx.x);
      if (kk + 64 < C_)
        __builtin_prefetch(xb + (size_t)(kk + 64) * P_ + p0, 0, 1);
    }

    // A fragment (16-bit A 16x32): lane m = lane&15,
    // elements 0-15 = K 16*(lane>>4) .. 16*(lane>>4)+15 (matches B)
    v8bf alo = *(const v8bf*)&lds_a[cur][m][khalf * 16];
    v8bf ahi = *(const v8bf*)&lds_a[cur][m][khalf * 16 + 8];
    v16bf af;
    #pragma unroll
    for (int e = 0; e < 8; ++e) { af[e] = alo[e]; af[e + 8] = ahi[e]; }

    // B fragments (32x16): lane n = n0+(lane&15), K = kk + 16*(lane>>4) + e
    const __bf16* wra = w1b + (size_t)na * C_ + kk + khalf * 16;
    const __bf16* wrb = w1b + (size_t)nb * C_ + kk + khalf * 16;
    v8bf b0lo = *(const v8bf*)wra; v8bf b0hi = *(const v8bf*)(wra + 8);
    v8bf b1lo = *(const v8bf*)wrb; v8bf b1hi = *(const v8bf*)(wrb + 8);
    v16bf bf0, bf1;
    #pragma unroll
    for (int e = 0; e < 8; ++e) {
      bf0[e] = b0lo[e]; bf0[e + 8] = b0hi[e];
      bf1[e] = b1lo[e]; bf1[e + 8] = b1hi[e];
    }

    acc0 = __builtin_amdgcn_wmma_f32_16x16x32_bf16(false, af, false, bf0,
                                                   (short)0, acc0, false, false);
    acc1 = __builtin_amdgcn_wmma_f32_16x16x32_bf16(false, af, false, bf1,
                                                   (short)0, acc1, false, false);

    __syncthreads();  // next-buffer writes done; cur reads done, restaged next iter
  }

  // Epilogue: BN + hardswish, mask invalid p, sum over this tile's rows.
  const float sa = g1[na] * rsqrtf(v1[na] + EPS_);
  const float sb = g1[nb] * rsqrtf(v1[nb] + EPS_);
  const float ma = m1[na], ba = b1[na];
  const float mb = m1[nb], bb = b1[nb];

  float sum0 = 0.f, sum1 = 0.f;
  const int mbase = khalf * 8;
  #pragma unroll
  for (int v = 0; v < 8; ++v) {
    int p = p0 + mbase + v;
    if (p < P_) {
      float h0 = (acc0[v] - ma) * sa + ba;
      sum0 += h0 * fminf(fmaxf(h0 + 3.f, 0.f), 6.f) * (1.f / 6.f);
      float h1 = (acc1[v] - mb) * sb + bb;
      sum1 += h1 * fminf(fmaxf(h1 + 3.f, 0.f), 6.f) * (1.f / 6.f);
    }
  }
  // combine the two m-halves (lane and lane^16 hold same n)
  sum0 += __shfl_xor(sum0, 16);
  sum1 += __shfl_xor(sum1, 16);
  if (lane < 16) {
    float* part = partial + ((size_t)b * MT_ + tile) * R_;
    part[na] = sum0;
    part[nb] = sum1;
  }
}

// ---------------------------------------------------------- K2: reduce -> mean
__global__ void k2_reduce_pool(const float* __restrict__ partial, float* __restrict__ pooled) {
  int b = blockIdx.x, r = threadIdx.x;  // 128 x 256
  float s = 0.f;
  #pragma unroll
  for (int t = 0; t < MT_; ++t) s += partial[((size_t)b * MT_ + t) * R_ + r];
  pooled[(size_t)b * R_ + r] = s * (1.0f / (float)P_);
}

// ------------------------------------------------------------------ K3: head
__global__ void k3_head(const float* __restrict__ pooled, const float* __restrict__ w2,
                        const float* __restrict__ g2, const float* __restrict__ b2,
                        const float* __restrict__ m2, const float* __restrict__ v2,
                        float* __restrict__ roi) {
  int b = threadIdx.x;  // 128 threads, 1 block
  const float SC[3] = {7.f, 7.f, 4.f};
  float sig[6];
  #pragma unroll
  for (int j = 0; j < 6; ++j) {
    float s = 0.f;
    const float* pr = pooled + (size_t)b * R_;
    const float* wr = w2 + (size_t)j * R_;
    for (int r = 0; r < R_; ++r) s += pr[r] * wr[r];
    s = (s - m2[j]) * (g2[j] * rsqrtf(v2[j] + EPS_)) + b2[j];
    sig[j] = 1.f / (1.f + __expf(-s));
  }
  #pragma unroll
  for (int d = 0; d < 3; ++d) {
    roi[b * 8 + d]     = 0.5f * sig[d] * SC[d];              // start (x,y,t)
    roi[b * 8 + 3 + d] = (1.f - 0.5f * sig[3 + d]) * SC[d];  // end (x,y,t)
  }
}

// --------------------------------------------------------------- K4: roi_align
__device__ __forceinline__ void prep1(float c, float size, int isize,
                                      int& lo, int& hi, float& w0, float& w1) {
  float valid = (c >= -1.f && c <= size) ? 1.f : 0.f;
  c = fminf(fmaxf(c, 0.f), size - 1.f);
  lo = (int)floorf(c);
  hi = min(lo + 1, isize - 1);
  float f = c - (float)lo;
  w0 = (1.f - f) * valid;
  w1 = f * valid;
}

// grid: (C_/64, B_), block 256. LDS-stages x[b, c0..c0+63, :] (contiguous 50KB).
__global__ void __launch_bounds__(256)
k4_roi(const float* __restrict__ x, const float* __restrict__ roi,
       float* __restrict__ out) {
  __shared__ __align__(16) float tile[64 * P_];  // 50176 B
  const int c0 = blockIdx.x * 64;
  const int b  = blockIdx.y;
  const float* src = x + ((size_t)b * C_ + c0) * P_;  // contiguous 12544 floats

#if USE_ASYNC_LDS
  for (int i = threadIdx.x; i < (64 * P_) / 4; i += 256) {
    __builtin_amdgcn_global_load_async_to_lds_b128(
        (__attribute__((address_space(1))) v4i*)(src + 4 * i),
        (__attribute__((address_space(3))) v4i*)(&tile[4 * i]), 0, 0);
  }
  wait_async0();
  __syncthreads();
#else
  for (int i = threadIdx.x; i < 64 * P_; i += 256) tile[i] = src[i];
  __syncthreads();
#endif

  const float s0 = roi[b * 8 + 0], s1 = roi[b * 8 + 1], s2 = roi[b * 8 + 2];
  const float e0 = roi[b * 8 + 3], e1 = roi[b * 8 + 4], e2 = roi[b * 8 + 5];

  for (int i = threadIdx.x; i < 64 * BINS_; i += 256) {
    const int cl  = i / BINS_;
    const int bin = i % BINS_;
    const int tt = bin / 25, rem = bin % 25, yy = rem / 5, xx = rem % 5;

    float tc = s2 + ((float)tt + 0.5f) * (e2 - s2) * (1.f / (float)OT_);
    float yc = s1 + ((float)yy + 0.5f) * (e1 - s1) * (1.f / (float)OH_);
    float xc = s0 + ((float)xx + 0.5f) * (e0 - s0) * (1.f / (float)OW_);

    int ti0, ti1, yi0, yi1, xi0, xi1;
    float tw0, tw1, yw0, yw1, xw0, xw1;
    prep1(tc, (float)T_, T_, ti0, ti1, tw0, tw1);
    prep1(yc, (float)H_, H_, yi0, yi1, yw0, yw1);
    prep1(xc, (float)W_, W_, xi0, xi1, xw0, xw1);

    const float* ct = &tile[cl * P_];
    float acc = 0.f;
    acc += ct[ti0 * 49 + yi0 * 7 + xi0] * (tw0 * yw0 * xw0);
    acc += ct[ti0 * 49 + yi0 * 7 + xi1] * (tw0 * yw0 * xw1);
    acc += ct[ti0 * 49 + yi1 * 7 + xi0] * (tw0 * yw1 * xw0);
    acc += ct[ti0 * 49 + yi1 * 7 + xi1] * (tw0 * yw1 * xw1);
    acc += ct[ti1 * 49 + yi0 * 7 + xi0] * (tw1 * yw0 * xw0);
    acc += ct[ti1 * 49 + yi0 * 7 + xi1] * (tw1 * yw0 * xw1);
    acc += ct[ti1 * 49 + yi1 * 7 + xi0] * (tw1 * yw1 * xw0);
    acc += ct[ti1 * 49 + yi1 * 7 + xi1] * (tw1 * yw1 * xw1);

    out[((size_t)b * C_ + c0 + cl) * BINS_ + bin] = acc;
  }
}

// ---------------------------------------------------------------------------
extern "C" void kernel_launch(void* const* d_in, const int* in_sizes, int n_in,
                              void* d_out, int out_size, void* d_ws, size_t ws_size,
                              hipStream_t stream) {
  const float* x  = (const float*)d_in[0];
  const float* w1 = (const float*)d_in[1];
  const float* g1 = (const float*)d_in[2];
  const float* b1 = (const float*)d_in[3];
  const float* m1 = (const float*)d_in[4];
  const float* v1 = (const float*)d_in[5];
  const float* w2 = (const float*)d_in[6];
  const float* g2 = (const float*)d_in[7];
  const float* b2 = (const float*)d_in[8];
  const float* m2 = (const float*)d_in[9];
  const float* v2 = (const float*)d_in[10];
  float* out = (float*)d_out;

  // workspace layout (needs ~2.36 MB)
  char* wsb = (char*)d_ws;
  __bf16* w1b    = (__bf16*)wsb;                               // 524288 B
  float*  partial = (float*)(wsb + 524288);                    // 1703936 B
  float*  pooled  = (float*)(wsb + 524288 + 1703936);          // 131072 B
  float*  roi     = (float*)(wsb + 524288 + 1703936 + 131072); // 4096 B

  k0_convert_w1<<<(R_ * C_) / 256, 256, 0, stream>>>(w1, w1b);
  k1_gemm_partial<<<dim3(MT_, B_), 256, 0, stream>>>(x, w1b, g1, b1, m1, v1, partial);
  k2_reduce_pool<<<B_, R_, 0, stream>>>(partial, pooled);
  k3_head<<<1, B_, 0, stream>>>(pooled, w2, g2, b2, m2, v2, roi);
  k4_roi<<<dim3(C_ / 64, B_), 256, 0, stream>>>(x, roi, out);
}
